// Llama4TextMoe_3968549782065
// MI455X (gfx1250) — matrix operand order repaired and checked by reading, not compiled, not run on hardware
//
#include <hip/hip_runtime.h>
#include <hip/hip_bf16.h>
#include <stdint.h>

// Problem dims (fixed by the reference).
#define T_TOK 2048   // B*S
#define HDIM  2048
#define IDIM  4096
#define NEXP  8

typedef __attribute__((ext_vector_type(16))) __bf16    v16bf;
typedef __attribute__((ext_vector_type(8)))  float     v8f;
typedef __attribute__((ext_vector_type(4)))  unsigned  u32x4;

struct Pack32 { u32x4 a, b; };

__device__ __forceinline__ unsigned short f32_to_bf16(float f) {
  unsigned u = __float_as_uint(f);
  u += 0x7FFFu + ((u >> 16) & 1u);          // round-to-nearest-even
  return (unsigned short)(u >> 16);
}

// Load 16 bf16 (two 16B chunks) into the WMMA per-lane register shape.
__device__ __forceinline__ v16bf load_bf16x16(const unsigned short* row, int o0, int o1) {
  Pack32 p;
  p.a = *reinterpret_cast<const u32x4*>(row + o0);
  p.b = *reinterpret_cast<const u32x4*>(row + o1);
  return __builtin_bit_cast(v16bf, p);
}

__device__ __forceinline__ float wave_sum(float v) {
#pragma unroll
  for (int off = 16; off > 0; off >>= 1) v += __shfl_down(v, off, 32);
  return v;
}

// ---------------------------------------------------------------- router ----
// One block per token; wave e computes dot(hs[t], router_w[e]); thread 0 does
// top-1 + sigmoid masking and writes router_scores[e*T + t].
__global__ void __launch_bounds__(256) k_router(const float* __restrict__ hs,
                                                const float* __restrict__ rw,
                                                float* __restrict__ scores) {
  const int t = blockIdx.x;
  const int w = threadIdx.x >> 5, lane = threadIdx.x & 31;
  const float4* x = reinterpret_cast<const float4*>(hs + (size_t)t * HDIM);
  const float4* r = reinterpret_cast<const float4*>(rw + (size_t)w * HDIM);
  float acc = 0.f;
#pragma unroll 4
  for (int k = lane; k < HDIM / 4; k += 32) {
    float4 a = x[k], b = r[k];
    acc += a.x * b.x + a.y * b.y + a.z * b.z + a.w * b.w;
  }
  acc = wave_sum(acc);
  __shared__ float slog[NEXP];
  if (lane == 0) slog[w] = acc;
  __syncthreads();
  if (threadIdx.x == 0) {
    int am = 0; float mv = slog[0];
#pragma unroll
    for (int e = 1; e < NEXP; ++e) if (slog[e] > mv) { mv = slog[e]; am = e; }
#pragma unroll
    for (int e = 0; e < NEXP; ++e) {
      float s = (e == am) ? 1.f / (1.f + __expf(-slog[e])) : 0.f;
      scores[(size_t)e * T_TOK + t] = s;
    }
  }
}

// ------------------------------------------------------ routed experts ------
// rows[e] = hs[token e] * scores.flat[e]; gate/up GEMVs (bandwidth-bound).
__global__ void __launch_bounds__(256) k_expert_gu(const float* __restrict__ hs,
                                                   const float* __restrict__ gw,
                                                   const float* __restrict__ uw,
                                                   const float* __restrict__ scores,
                                                   float* __restrict__ act_exp) {
  const int w = threadIdx.x >> 5, lane = threadIdx.x & 31;
  const int i = blockIdx.x * 8 + w;
  const int e = blockIdx.y;
  const float4* x = reinterpret_cast<const float4*>(hs + (size_t)e * HDIM);
  const float4* g = reinterpret_cast<const float4*>(gw + ((size_t)e * IDIM + i) * HDIM);
  const float4* u = reinterpret_cast<const float4*>(uw + ((size_t)e * IDIM + i) * HDIM);
  float dg = 0.f, du = 0.f;
#pragma unroll 4
  for (int k = lane; k < HDIM / 4; k += 32) {
    float4 a = x[k], bg = g[k], bu = u[k];
    dg += a.x * bg.x + a.y * bg.y + a.z * bg.z + a.w * bg.w;
    du += a.x * bu.x + a.y * bu.y + a.z * bu.z + a.w * bu.w;
  }
  dg = wave_sum(dg);
  du = wave_sum(du);
  if (lane == 0) {
    float sc = scores[e];            // router_scores.flat[:E] == scores[0][0:E]
    float gg = sc * dg, uu = sc * du;
    act_exp[(size_t)e * IDIM + i] = gg / (1.f + __expf(-gg)) * uu;   // silu(g)*u
  }
}

__global__ void __launch_bounds__(256) k_expert_down(const float* __restrict__ act_exp,
                                                     const float* __restrict__ dw,
                                                     float* __restrict__ eo) {
  const int w = threadIdx.x >> 5, lane = threadIdx.x & 31;
  const int h = blockIdx.x * 8 + w;
  const int e = blockIdx.y;
  const float4* a = reinterpret_cast<const float4*>(act_exp + (size_t)e * IDIM);
  const float4* d = reinterpret_cast<const float4*>(dw + ((size_t)e * HDIM + h) * IDIM);
  float acc = 0.f;
#pragma unroll 4
  for (int k = lane; k < IDIM / 4; k += 32) {
    float4 x = a[k], y = d[k];
    acc += x.x * y.x + x.y * y.y + x.z * y.z + x.w * y.w;
  }
  acc = wave_sum(acc);
  if (lane == 0) eo[(size_t)e * HDIM + h] = acc;
}

__global__ void __launch_bounds__(256) k_expert_bias(const float* __restrict__ eo,
                                                     float* __restrict__ bias) {
  int h = blockIdx.x * blockDim.x + threadIdx.x;
  if (h < HDIM) {
    float s = 0.f;
#pragma unroll
    for (int e = 0; e < NEXP; ++e) s += eo[(size_t)e * HDIM + h];
    bias[h] = s;                     // deterministic sequential sum, no atomics
  }
}

// ----------------------------------------------------- fp32 -> bf16 --------
__global__ void __launch_bounds__(256) k_cvt_bf16(const float* __restrict__ in,
                                                  unsigned short* __restrict__ out,
                                                  int n) {
  for (int i = blockIdx.x * blockDim.x + threadIdx.x; i < n; i += gridDim.x * blockDim.x)
    out[i] = f32_to_bf16(in[i]);
}

// ------------------------------------------------ shared MLP, stage 1 ------
// G = X@Wg^T, U = X@Wu^T, Act = silu(G)*U  (bf16 WMMA, fp32 accumulate).
// Wave computes a 64x32 tile of both G and U: 16 WMMA per 16 b128-loads per
// K=32 step (1.0 loads/WMMA), C=128 + A=32 + B=32 VGPRs.
#define WMMA_BF16(A, B, C) \
  __builtin_amdgcn_wmma_f32_16x16x32_bf16(false, (A), false, (B), (short)0, (C), false, false)

__global__ void __launch_bounds__(256) k_mlp1(const unsigned short* __restrict__ X,
                                              const unsigned short* __restrict__ Wg,
                                              const unsigned short* __restrict__ Wu,
                                              unsigned short* __restrict__ Act) {
  const int NT = IDIM / 32;                                  // 128 N-tiles
  const int gw = blockIdx.x * 8 + (threadIdx.x >> 5);
  const int m0 = (gw / NT) * 64, n0 = (gw % NT) * 32;
  const int lane = threadIdx.x & 31, lrow = lane & 15, lh = lane >> 4;

  const unsigned short* ap[4];
#pragma unroll
  for (int mi = 0; mi < 4; ++mi) ap[mi] = X + (size_t)(m0 + 16 * mi + lrow) * HDIM;
  const unsigned short* gp[2];
  const unsigned short* up[2];
#pragma unroll
  for (int ni = 0; ni < 2; ++ni) {
    gp[ni] = Wg + (size_t)(n0 + 16 * ni + lrow) * HDIM;
    up[ni] = Wu + (size_t)(n0 + 16 * ni + lrow) * HDIM;
  }

  v8f cg[4][2] = {};
  v8f cu[4][2] = {};

  for (int kk = 0; kk < HDIM; kk += 32) {
    const int ao0 = kk + lh * 16, ao1 = kk + lh * 16 + 8;    // A: contiguous 16 per half
    const int bo  = kk + lh * 16;                            // B: contiguous 16 per half
    v16bf A[4], Bg[2], Bu[2];
#pragma unroll
    for (int mi = 0; mi < 4; ++mi) A[mi] = load_bf16x16(ap[mi], ao0, ao1);
#pragma unroll
    for (int ni = 0; ni < 2; ++ni) {
      Bg[ni] = load_bf16x16(gp[ni], bo, bo + 8);
      Bu[ni] = load_bf16x16(up[ni], bo, bo + 8);
    }
#pragma unroll
    for (int mi = 0; mi < 4; ++mi)
#pragma unroll
      for (int ni = 0; ni < 2; ++ni) {
        cg[mi][ni] = WMMA_BF16(A[mi], Bg[ni], cg[mi][ni]);
        cu[mi][ni] = WMMA_BF16(A[mi], Bu[ni], cu[mi][ni]);
      }
  }

#pragma unroll
  for (int mi = 0; mi < 4; ++mi)
#pragma unroll
    for (int ni = 0; ni < 2; ++ni)
#pragma unroll
      for (int r = 0; r < 8; ++r) {
        const int row = m0 + mi * 16 + r + lh * 8;           // C/D: M = r + 8*(lane>=16)
        const int col = n0 + ni * 16 + lrow;
        float gv = cg[mi][ni][r], uv = cu[mi][ni][r];
        float av = gv / (1.f + __expf(-gv)) * uv;
        Act[(size_t)row * IDIM + col] = f32_to_bf16(av);
      }
}

// ------------------------------------------------ shared MLP, stage 2 ------
// Out = Act@Wd^T + expert_bias (broadcast over rows), fp32 output.
// Wave computes 64x64: 16 WMMA per 16 loads per K=32 step.
__global__ void __launch_bounds__(256) k_mlp2(const unsigned short* __restrict__ Act,
                                              const unsigned short* __restrict__ Wd,
                                              const float* __restrict__ bias,
                                              float* __restrict__ out) {
  const int NT = HDIM / 64;                                  // 32 N-tiles
  const int gw = blockIdx.x * 8 + (threadIdx.x >> 5);
  const int m0 = (gw / NT) * 64, n0 = (gw % NT) * 64;
  const int lane = threadIdx.x & 31, lrow = lane & 15, lh = lane >> 4;

  const unsigned short* ap[4];
  const unsigned short* bp[4];
#pragma unroll
  for (int i = 0; i < 4; ++i) {
    ap[i] = Act + (size_t)(m0 + 16 * i + lrow) * IDIM;
    bp[i] = Wd  + (size_t)(n0 + 16 * i + lrow) * IDIM;
  }

  v8f c[4][4] = {};

  for (int kk = 0; kk < IDIM; kk += 32) {
    const int ao0 = kk + lh * 16, ao1 = kk + lh * 16 + 8;
    const int bo  = kk + lh * 16;
    v16bf A[4], B[4];
#pragma unroll
    for (int i = 0; i < 4; ++i) {
      A[i] = load_bf16x16(ap[i], ao0, ao1);
      B[i] = load_bf16x16(bp[i], bo, bo + 8);
    }
#pragma unroll
    for (int mi = 0; mi < 4; ++mi)
#pragma unroll
      for (int ni = 0; ni < 4; ++ni)
        c[mi][ni] = WMMA_BF16(A[mi], B[ni], c[mi][ni]);
  }

#pragma unroll
  for (int mi = 0; mi < 4; ++mi)
#pragma unroll
    for (int ni = 0; ni < 4; ++ni)
#pragma unroll
      for (int r = 0; r < 8; ++r) {
        const int row = m0 + mi * 16 + r + lh * 8;
        const int col = n0 + ni * 16 + lrow;
        out[(size_t)row * HDIM + col] = c[mi][ni][r] + bias[col];
      }
}

// ---------------------------------------------------------------------------
extern "C" void kernel_launch(void* const* d_in, const int* in_sizes, int n_in,
                              void* d_out, int out_size, void* d_ws, size_t ws_size,
                              hipStream_t stream) {
  (void)in_sizes; (void)n_in; (void)out_size; (void)ws_size;
  const float* hs  = (const float*)d_in[0];
  const float* rw  = (const float*)d_in[1];
  const float* sgw = (const float*)d_in[2];
  const float* suw = (const float*)d_in[3];
  const float* sdw = (const float*)d_in[4];
  const float* egw = (const float*)d_in[5];
  const float* euw = (const float*)d_in[6];
  const float* edw = (const float*)d_in[7];

  float* out_main = (float*)d_out;                           // [T, H]
  float* scores   = out_main + (size_t)T_TOK * HDIM;         // [E, T]

  // Workspace layout (~72.2 MB total).
  float* act_exp = (float*)d_ws;                             // E*I
  float* eo      = act_exp + (size_t)NEXP * IDIM;            // E*H
  float* bias    = eo + (size_t)NEXP * HDIM;                 // H
  unsigned short* Xbf  = (unsigned short*)(bias + HDIM);     // T*H bf16
  unsigned short* Wgbf = Xbf  + (size_t)T_TOK * HDIM;        // I*H
  unsigned short* Wubf = Wgbf + (size_t)IDIM * HDIM;         // I*H
  unsigned short* Wdbf = Wubf + (size_t)IDIM * HDIM;         // H*I
  unsigned short* Actb = Wdbf + (size_t)HDIM * IDIM;         // T*I

  // Router + expert path (bandwidth-bound GEMVs).
  k_router<<<T_TOK, 256, 0, stream>>>(hs, rw, scores);
  k_expert_gu<<<dim3(IDIM / 8, NEXP), 256, 0, stream>>>(hs, egw, euw, scores, act_exp);
  k_expert_down<<<dim3(HDIM / 8, NEXP), 256, 0, stream>>>(act_exp, edw, eo);
  k_expert_bias<<<HDIM / 256, 256, 0, stream>>>(eo, bias);

  // fp32 -> bf16 for the WMMA path.
  k_cvt_bf16<<<2048, 256, 0, stream>>>(hs,  Xbf,  T_TOK * HDIM);
  k_cvt_bf16<<<2048, 256, 0, stream>>>(sgw, Wgbf, IDIM * HDIM);
  k_cvt_bf16<<<2048, 256, 0, stream>>>(suw, Wubf, IDIM * HDIM);
  k_cvt_bf16<<<2048, 256, 0, stream>>>(sdw, Wdbf, HDIM * IDIM);

  // Shared SwiGLU MLP with bf16 WMMA.
  k_mlp1<<<(T_TOK / 64) * (IDIM / 32) / 8, 256, 0, stream>>>(Xbf, Wgbf, Wubf, Actb);
  k_mlp2<<<(T_TOK / 64) * (HDIM / 64) / 8, 256, 0, stream>>>(Actb, Wdbf, bias, out_main);
}
